// HeposMultiHeadAttention_36593121362394
// MI455X (gfx1250) — matrix operand order repaired and checked by reading, not compiled, not run on hardware
//
#include <hip/hip_runtime.h>
#include <stdint.h>

// Problem constants (match reference)
#define DMODEL 1024
#define NHEAD  16
#define HDIM   64
#define BATCH  4
#define TQLEN  2048
#define TKLEN  8192
#define LLEN   512   // TK / STRIDE

typedef __attribute__((ext_vector_type(16))) __bf16 v16bf;
typedef __attribute__((ext_vector_type(8)))  float  v8f;
typedef __attribute__((ext_vector_type(4)))  unsigned int v4u;
typedef __attribute__((ext_vector_type(4)))  int    v4i;
typedef __attribute__((ext_vector_type(8)))  int    v8i;

union BF16Frag { v16bf v; uint4 q[2]; };

__device__ __forceinline__ uint16_t f2bf(float f) {
    uint32_t u = __builtin_bit_cast(uint32_t, f);
    return (uint16_t)((u + 0x7FFFu + ((u >> 16) & 1u)) >> 16);   // RNE
}
__device__ __forceinline__ uint32_t pack2(float a, float b) {
    return (uint32_t)f2bf(a) | ((uint32_t)f2bf(b) << 16);
}
__device__ __forceinline__ v8f wmma_bf16(v16bf a, v16bf b, v8f c) {
    return __builtin_amdgcn_wmma_f32_16x16x32_bf16(false, a, false, b, (short)0, c,
                                                   false, false);
}

// ---------------------------------------------------------------------------
// f32 -> bf16 conversion (weights)
// ---------------------------------------------------------------------------
__global__ __launch_bounds__(256) void cvt_f32_to_bf16(const float* __restrict__ s,
                                                       uint16_t* __restrict__ d, int n4) {
    int i = blockIdx.x * 256 + threadIdx.x;
    if (i < n4) {
        float4 v = ((const float4*)s)[i];
        uint2 o; o.x = pack2(v.x, v.y); o.y = pack2(v.z, v.w);
        ((uint2*)d)[i] = o;
    }
}

// ---------------------------------------------------------------------------
// GEMM: Out[m][n] = sum_d A[m][d] * W[n][d] + bias[n]
//   A: f32 (converted on the fly) or bf16; W: bf16; Out: bf16 or f32.
//   Block tile 128x128, 8 waves (4Mx2N), each wave 32x64 via 2x4 C-frags.
//   K-step = 32, double-buffered LDS, rows padded to 40 halves.
// ---------------------------------------------------------------------------
template <bool A_F32, bool OUT_BF16>
__global__ __launch_bounds__(256) void gemm_wt(const void* __restrict__ Ap,
                                               const uint16_t* __restrict__ W,
                                               const float* __restrict__ bias,
                                               void* __restrict__ Outp) {
    __shared__ __align__(16) uint16_t As[2][128][40];
    __shared__ __align__(16) uint16_t Bs[2][128][40];

    const int tid  = threadIdx.x;
    const int lane = tid & 31;
    const int wid  = tid >> 5;
    const int half = lane >> 4;
    const int ln   = lane & 15;
    const int wm   = wid & 3;    // 32-row strip
    const int wn   = wid >> 2;   // 64-col strip
    const long gm0 = (long)blockIdx.x * 128;
    const long gn0 = (long)blockIdx.y * 128;
    const int lrow = tid >> 1;        // 0..127 (tile row this thread fetches)
    const int lcol = (tid & 1) << 4;  // 0 or 16 within the 32-wide k-step

    const float*    Af = (const float*)Ap;
    const uint16_t* Ab = (const uint16_t*)Ap;

    v8f acc[2][4];
#pragma unroll
    for (int i = 0; i < 2; ++i)
#pragma unroll
        for (int j = 0; j < 4; ++j)
#pragma unroll
            for (int r = 0; r < 8; ++r) acc[i][j][r] = 0.0f;

    // prologue: k-step 0 -> buffer 0
    {
        if (A_F32) {
            const float4* pa = (const float4*)(Af + (gm0 + lrow) * DMODEL + lcol);
            float4 t0 = pa[0], t1 = pa[1], t2 = pa[2], t3 = pa[3];
            *(uint4*)&As[0][lrow][lcol] =
                uint4{pack2(t0.x, t0.y), pack2(t0.z, t0.w), pack2(t1.x, t1.y), pack2(t1.z, t1.w)};
            *(uint4*)&As[0][lrow][lcol + 8] =
                uint4{pack2(t2.x, t2.y), pack2(t2.z, t2.w), pack2(t3.x, t3.y), pack2(t3.z, t3.w)};
        } else {
            const uint4* pa = (const uint4*)(Ab + (gm0 + lrow) * DMODEL + lcol);
            *(uint4*)&As[0][lrow][lcol]     = pa[0];
            *(uint4*)&As[0][lrow][lcol + 8] = pa[1];
        }
        const uint4* pw = (const uint4*)(W + (gn0 + lrow) * DMODEL + lcol);
        *(uint4*)&Bs[0][lrow][lcol]     = pw[0];
        *(uint4*)&Bs[0][lrow][lcol + 8] = pw[1];
    }
    __syncthreads();

    float4 t0, t1, t2, t3;
    uint4  ua0, ua1, uw0, uw1;
#pragma unroll 1
    for (int kt = 0; kt < DMODEL / 32; ++kt) {
        const int  buf  = kt & 1;
        const bool more = (kt + 1 < DMODEL / 32);
        if (more) {  // prefetch next k-step into registers
            const int k0 = (kt + 1) * 32;
            if (A_F32) {
                const float4* pa = (const float4*)(Af + (gm0 + lrow) * DMODEL + k0 + lcol);
                t0 = pa[0]; t1 = pa[1]; t2 = pa[2]; t3 = pa[3];
            } else {
                const uint4* pa = (const uint4*)(Ab + (gm0 + lrow) * DMODEL + k0 + lcol);
                ua0 = pa[0]; ua1 = pa[1];
            }
            const uint4* pw = (const uint4*)(W + (gn0 + lrow) * DMODEL + k0 + lcol);
            uw0 = pw[0]; uw1 = pw[1];
        }
        // A-frags: lane<16 -> K 0-7 & 16-23 of row ln; lane>=16 -> K 8-15 & 24-31
        BF16Frag a0, a1;
        a0.q[0] = *(const uint4*)&As[buf][wm * 32 + ln][8 * half];
        a0.q[1] = *(const uint4*)&As[buf][wm * 32 + ln][16 + 8 * half];
        a1.q[0] = *(const uint4*)&As[buf][wm * 32 + 16 + ln][8 * half];
        a1.q[1] = *(const uint4*)&As[buf][wm * 32 + 16 + ln][16 + 8 * half];
#pragma unroll
        for (int j = 0; j < 4; ++j) {
            // B-frag: lane<16 -> K 0-7 & 16-23 of column ln; lane>=16 -> K 8-15 & 24-31
            BF16Frag bj;
            bj.q[0] = *(const uint4*)&Bs[buf][wn * 64 + 16 * j + ln][8 * half];
            bj.q[1] = *(const uint4*)&Bs[buf][wn * 64 + 16 * j + ln][16 + 8 * half];
            acc[0][j] = wmma_bf16(a0.v, bj.v, acc[0][j]);
            acc[1][j] = wmma_bf16(a1.v, bj.v, acc[1][j]);
        }
        if (more) {
            const int nb = buf ^ 1;
            if (A_F32) {
                *(uint4*)&As[nb][lrow][lcol] =
                    uint4{pack2(t0.x, t0.y), pack2(t0.z, t0.w), pack2(t1.x, t1.y), pack2(t1.z, t1.w)};
                *(uint4*)&As[nb][lrow][lcol + 8] =
                    uint4{pack2(t2.x, t2.y), pack2(t2.z, t2.w), pack2(t3.x, t3.y), pack2(t3.z, t3.w)};
            } else {
                *(uint4*)&As[nb][lrow][lcol]     = ua0;
                *(uint4*)&As[nb][lrow][lcol + 8] = ua1;
            }
            *(uint4*)&Bs[nb][lrow][lcol]     = uw0;
            *(uint4*)&Bs[nb][lrow][lcol + 8] = uw1;
        }
        __syncthreads();
    }

    // epilogue: C-frag element r -> row r + 8*half, col ln
#pragma unroll
    for (int i = 0; i < 2; ++i)
#pragma unroll
        for (int j = 0; j < 4; ++j) {
            const long gn = gn0 + wn * 64 + 16 * j + ln;
            const float bv = bias[gn];
#pragma unroll
            for (int r = 0; r < 8; ++r) {
                const long gm = gm0 + wm * 32 + 16 * i + 8 * half + r;
                const float val = acc[i][j][r] + bv;
                if (OUT_BF16) ((uint16_t*)Outp)[gm * DMODEL + gn] = f2bf(val);
                else          ((float*)Outp)[gm * DMODEL + gn]    = val;
            }
        }
}

// ---------------------------------------------------------------------------
// Strided-head attention. One block per (b, h, 128-query tile).
// Head h uses K/V rows h+16*l (l=0..511), cols [h*64, h*64+64).
// K slice is fetched by the Tensor Data Mover (one DMA, hardware-padded rows);
// V slice is transposed into LDS by the ALUs in parallel with the DMA.
// Whole K + V^T slice resident in LDS (~157 KB of the 320 KB WGP LDS).
// 8 waves x 16 queries, flash-softmax over 8 L-tiles of 64.
// ---------------------------------------------------------------------------
__global__ __launch_bounds__(256) void attn_kernel(const uint16_t* __restrict__ Qb,
                                                   const uint16_t* __restrict__ Kb,
                                                   const uint16_t* __restrict__ Vb,
                                                   const int* __restrict__ mask,
                                                   uint16_t* __restrict__ Ctx) {
    __shared__ __align__(16) uint16_t Ks[LLEN][72];      // [l][d], 73.7 KB (TDM-padded)
    __shared__ __align__(16) uint16_t Vt[HDIM][520];     // [d][l], 66.6 KB
    __shared__ __align__(16) uint16_t Ps[8][16][72];     // per-wave P scratch
    __shared__ int Ms[LLEN];

    const int b = blockIdx.z, h = blockIdx.y, qblk = blockIdx.x;
    const int tid = threadIdx.x, lane = tid & 31, wid = tid >> 5;
    const int half = lane >> 4, ln = lane & 15;

    // ---- K slice -> LDS via TDM (wave 0 issues one descriptor) ----
    // 2D tile: 512 rows x 64 bf16, row stride 16*DMODEL elems; LDS rows padded
    // 128B data + 16B pad (pad_interval=4 -> 32 DWORDs, pad_amount=3 -> 4 DWORDs)
    // so the in-LDS pitch is exactly Ks's 72 halves.
    if (tid < 32) {
        const uint64_t gaddr =
            (uint64_t)(uintptr_t)(Kb + ((size_t)b * TKLEN + h) * DMODEL + h * HDIM);
        const uint32_t laddr = (uint32_t)(uintptr_t)&Ks[0][0];
        v4u g0;
        g0[0] = 1u;                                   // count=1 (valid user D#)
        g0[1] = laddr;                                // lds_addr
        g0[2] = (uint32_t)gaddr;                      // global_addr[31:0]
        g0[3] = (uint32_t)((gaddr >> 32) & 0x1FFFFFFu) | (2u << 30);  // [56:32], type=2
        v8i g1;
        g1[0] = (int)((1u << 16)          // data_size = 2 bytes
                    | (1u << 20)          // pad_enable
                    | (4u << 22)          // pad_interval: 32 DWORDs (=128 B)
                    | (3u << 25));        // pad_amount:   4 DWORDs (=16 B)
        g1[1] = (int)(64u << 16);         // tensor_dim0 = 64
        g1[2] = (int)(512u << 16);        // tensor_dim1 = 512
        g1[3] = (int)(64u << 16);         // tile_dim0 = 64
        g1[4] = 512;                      // tile_dim1 = 512
        g1[5] = 16 * DMODEL;              // tensor_dim0_stride = 16384 elems
        g1[6] = 0;
        g1[7] = 0;
        v4i gz = {0, 0, 0, 0};
#if __clang_major__ >= 23
        v8i gz8 = {0, 0, 0, 0, 0, 0, 0, 0};
        __builtin_amdgcn_tensor_load_to_lds(g0, g1, gz, gz, gz8, 0);
#else
        __builtin_amdgcn_tensor_load_to_lds(g0, g1, gz, gz, 0);
#endif
    }

    // ---- V slice -> LDS transposed (overlaps with the TDM) ----
    for (int i = tid; i < LLEN * 8; i += 256) {
        const int l = i >> 3, c = i & 7;
        uint4 dv = *(const uint4*)(Vb + ((size_t)b * TKLEN + h + 16 * l) * DMODEL + h * HDIM + c * 8);
        const uint16_t* e = (const uint16_t*)&dv;
#pragma unroll
        for (int j = 0; j < 8; ++j) Vt[c * 8 + j][l] = e[j];
    }
    for (int i = tid; i < LLEN; i += 256)
        Ms[i] = mask[(size_t)b * TKLEN + h + 16 * i];

    if (tid < 32) __builtin_amdgcn_s_wait_tensorcnt(0);  // K DMA done before barrier
    __syncthreads();

    // Q A-frags (16 queries per wave), loaded straight from global bf16
    const size_t qrow = (size_t)b * TQLEN + (size_t)qblk * 128 + wid * 16 + ln;
    const uint16_t* qp = Qb + qrow * DMODEL + h * HDIM;
    BF16Frag aq[2];
#pragma unroll
    for (int i = 0; i < 2; ++i) {
        aq[i].q[0] = *(const uint4*)(qp + 32 * i + 16 * half);
        aq[i].q[1] = *(const uint4*)(qp + 32 * i + 16 * half + 8);
    }

    v8f o[4];
    float rm[8], rl[8];
#pragma unroll
    for (int j = 0; j < 4; ++j)
#pragma unroll
        for (int r = 0; r < 8; ++r) o[j][r] = 0.0f;
#pragma unroll
    for (int r = 0; r < 8; ++r) { rm[r] = -1.0e30f; rl[r] = 0.0f; }

#pragma unroll 1
    for (int lt = 0; lt < 8; ++lt) {
        const int l0 = lt * 64;
        // S = Q @ K^T   (4 col-frags of 16, two K-depth steps over d)
        v8f s[4];
#pragma unroll
        for (int j = 0; j < 4; ++j) {
#pragma unroll
            for (int r = 0; r < 8; ++r) s[j][r] = 0.0f;
            const uint16_t* kp = &Ks[l0 + 16 * j + ln][0];
            BF16Frag b0, b1;
            b0.q[0] = *(const uint4*)(kp + 16 * half);
            b0.q[1] = *(const uint4*)(kp + 16 * half + 8);
            b1.q[0] = *(const uint4*)(kp + 32 + 16 * half);
            b1.q[1] = *(const uint4*)(kp + 32 + 16 * half + 8);
            s[j] = wmma_bf16(aq[0].v, b0.v, s[j]);
            s[j] = wmma_bf16(aq[1].v, b1.v, s[j]);
        }
        // scale 1/sqrt(64) + mask
#pragma unroll
        for (int j = 0; j < 4; ++j) {
            const int mv = Ms[l0 + 16 * j + ln];
#pragma unroll
            for (int r = 0; r < 8; ++r)
                s[j][r] = mv ? s[j][r] * 0.125f : -1.0e9f;
        }
        // flash-softmax update (row = r + 8*half lives in one 16-lane half)
#pragma unroll
        for (int r = 0; r < 8; ++r) {
            float m = fmaxf(fmaxf(s[0][r], s[1][r]), fmaxf(s[2][r], s[3][r]));
            m = fmaxf(m, __shfl_xor(m, 1));
            m = fmaxf(m, __shfl_xor(m, 2));
            m = fmaxf(m, __shfl_xor(m, 4));
            m = fmaxf(m, __shfl_xor(m, 8));
            const float nm   = fmaxf(rm[r], m);
            const float corr = __expf(rm[r] - nm);
            rm[r] = nm;
            float rs = 0.0f;
#pragma unroll
            for (int j = 0; j < 4; ++j) {
                const float p = __expf(s[j][r] - nm);
                s[j][r] = p;
                rs += p;
            }
            rs += __shfl_xor(rs, 1);
            rs += __shfl_xor(rs, 2);
            rs += __shfl_xor(rs, 4);
            rs += __shfl_xor(rs, 8);
            rl[r] = rl[r] * corr + rs;
#pragma unroll
            for (int j = 0; j < 4; ++j) o[j][r] *= corr;
        }
        // P: C-layout -> LDS -> A-layout (cross-lane, needs explicit dscnt wait)
#pragma unroll
        for (int j = 0; j < 4; ++j)
#pragma unroll
            for (int r = 0; r < 8; ++r)
                Ps[wid][r + 8 * half][16 * j + ln] = f2bf(s[j][r]);
        asm volatile("s_wait_dscnt 0" ::: "memory");
        BF16Frag ap[2];
        const uint16_t* pp = &Ps[wid][ln][0];
#pragma unroll
        for (int i = 0; i < 2; ++i) {
            ap[i].q[0] = *(const uint4*)(pp + 32 * i + 16 * half);
            ap[i].q[1] = *(const uint4*)(pp + 32 * i + 16 * half + 8);
        }
        // O += P @ V  (B-frags from transposed V: contiguous in l)
#pragma unroll
        for (int j = 0; j < 4; ++j) {
            const uint16_t* vp = &Vt[16 * j + ln][l0];
            BF16Frag b0, b1;
            b0.q[0] = *(const uint4*)(vp + 16 * half);
            b0.q[1] = *(const uint4*)(vp + 16 * half + 8);
            b1.q[0] = *(const uint4*)(vp + 32 + 16 * half);
            b1.q[1] = *(const uint4*)(vp + 32 + 16 * half + 8);
            o[j] = wmma_bf16(ap[0].v, b0.v, o[j]);
            o[j] = wmma_bf16(ap[1].v, b1.v, o[j]);
        }
    }

    // normalize + write context (bf16)
#pragma unroll
    for (int j = 0; j < 4; ++j)
#pragma unroll
        for (int r = 0; r < 8; ++r) {
            const float  val = o[j][r] / rl[r];
            const size_t qg  = (size_t)b * TQLEN + (size_t)qblk * 128 + wid * 16 + 8 * half + r;
            Ctx[qg * DMODEL + h * HDIM + 16 * j + ln] = f2bf(val);
        }
}

// ---------------------------------------------------------------------------
extern "C" void kernel_launch(void* const* d_in, const int* in_sizes, int n_in,
                              void* d_out, int out_size, void* d_ws, size_t ws_size,
                              hipStream_t stream) {
    (void)in_sizes; (void)n_in; (void)out_size; (void)ws_size;
    const float* q    = (const float*)d_in[0];
    const float* k    = (const float*)d_in[1];
    const float* v    = (const float*)d_in[2];
    const int*   msk  = (const int*)d_in[3];
    const float* w_q  = (const float*)d_in[4];
    const float* b_q  = (const float*)d_in[5];
    const float* w_k  = (const float*)d_in[6];
    const float* b_k  = (const float*)d_in[7];
    const float* w_v  = (const float*)d_in[8];
    const float* b_v  = (const float*)d_in[9];
    const float* w_o  = (const float*)d_in[10];
    const float* b_o  = (const float*)d_in[11];
    float* out = (float*)d_out;

    // workspace layout (bf16 units): 4 weights + Q + K + V + Ctx  (~168 MiB)
    uint16_t* wsp = (uint16_t*)d_ws;
    uint16_t* wqb = wsp;
    uint16_t* wkb = wqb + (size_t)DMODEL * DMODEL;
    uint16_t* wvb = wkb + (size_t)DMODEL * DMODEL;
    uint16_t* wob = wvb + (size_t)DMODEL * DMODEL;
    uint16_t* Qb  = wob + (size_t)DMODEL * DMODEL;
    uint16_t* Kb  = Qb  + (size_t)BATCH * TQLEN * DMODEL;
    uint16_t* Vb  = Kb  + (size_t)BATCH * TKLEN * DMODEL;
    uint16_t* Ctx = Vb  + (size_t)BATCH * TKLEN * DMODEL;

    const int n4 = (DMODEL * DMODEL) / 4;  // 262144
    cvt_f32_to_bf16<<<n4 / 256, 256, 0, stream>>>(w_q, wqb, n4);
    cvt_f32_to_bf16<<<n4 / 256, 256, 0, stream>>>(w_k, wkb, n4);
    cvt_f32_to_bf16<<<n4 / 256, 256, 0, stream>>>(w_v, wvb, n4);
    cvt_f32_to_bf16<<<n4 / 256, 256, 0, stream>>>(w_o, wob, n4);

    gemm_wt<true, true><<<dim3((BATCH * TQLEN) / 128, DMODEL / 128), 256, 0, stream>>>(q, wqb, b_q, Qb);
    gemm_wt<true, true><<<dim3((BATCH * TKLEN) / 128, DMODEL / 128), 256, 0, stream>>>(k, wkb, b_k, Kb);
    gemm_wt<true, true><<<dim3((BATCH * TKLEN) / 128, DMODEL / 128), 256, 0, stream>>>(v, wvb, b_v, Vb);

    attn_kernel<<<dim3(TQLEN / 128, NHEAD, BATCH), 256, 0, stream>>>(Qb, Kb, Vb, msk, Ctx);

    gemm_wt<false, false><<<dim3((BATCH * TQLEN) / 128, DMODEL / 128), 256, 0, stream>>>(Ctx, wob, b_o, out);
}
